// ChannelWiseCrossAttention_75539884802717
// MI455X (gfx1250) — matrix operand order repaired and checked by reading, not compiled, not run on hardware
//
#include <hip/hip_runtime.h>
#include <hip/hip_bf16.h>
#include <math.h>

#define BATCH 4
#define CCH   64
#define NPIX  4096   // 64*64

typedef __attribute__((ext_vector_type(16))) __bf16 v16bf;
typedef __attribute__((ext_vector_type(8)))  __bf16 v8bf;
typedef __attribute__((ext_vector_type(8)))  float  v8f;

static __device__ __forceinline__ __bf16 f2bf(float f) {
    return (__bf16)f;   // hardware cvt on gfx1250
}

// xor-16 lane exchange across the two 16-lane halves of a wave32 (pure VALU).
static __device__ __forceinline__ float xor16(float x) {
#if __has_builtin(__builtin_amdgcn_permlanex16)
    int xi = __builtin_bit_cast(int, x);
    int r  = __builtin_amdgcn_permlanex16(xi, xi, 0x76543210, 0xfedcba98, false, false);
    return __builtin_bit_cast(float, r);
#else
    return __shfl_xor(x, 16, 32);
#endif
}

// ---------------------------------------------------------------------------
// Kernel 1: q/k/v projections (1x1 conv). Memory-bound, writes bf16 operands
// in WMMA-friendly layouts: qT/kT as [b][pix][c], v as [b][c][pix].
// ---------------------------------------------------------------------------
__global__ void cwca_proj_kernel(const float* __restrict__ x1,
                                 const float* __restrict__ x2,
                                 const float* __restrict__ wq, const float* __restrict__ bq,
                                 const float* __restrict__ wk, const float* __restrict__ bk,
                                 const float* __restrict__ wv, const float* __restrict__ bv,
                                 __bf16* __restrict__ qT,
                                 __bf16* __restrict__ kT,
                                 __bf16* __restrict__ vM)
{
    int idx = blockIdx.x * blockDim.x + threadIdx.x;   // B*C*N threads
    int p  = idx & (NPIX - 1);
    int co = (idx >> 12) & (CCH - 1);
    int b  = idx >> 18;

    const float* xr1 = x1 + (size_t)b * CCH * NPIX + p;   // stride NPIX per ci
    const float* xr2 = x2 + (size_t)b * CCH * NPIX + p;
    const float* wqr = wq + (size_t)co * CCH;
    const float* wkr = wk + (size_t)co * CCH;
    const float* wvr = wv + (size_t)co * CCH;

    float accq = bq[co], acck = bk[co], accv = bv[co];
#pragma unroll 8
    for (int ci = 0; ci < CCH; ++ci) {
        float a1 = xr1[(size_t)ci * NPIX];
        float a2 = xr2[(size_t)ci * NPIX];
        accq = fmaf(wqr[ci], a1, accq);
        acck = fmaf(wkr[ci], a2, acck);
        accv = fmaf(wvr[ci], a2, accv);
    }
    qT[((size_t)b * NPIX + p) * CCH + co] = f2bf(accq);
    kT[((size_t)b * NPIX + p) * CCH + co] = f2bf(acck);
    vM[((size_t)b * CCH + co) * NPIX + p] = f2bf(accv);
}

// A-operand fragment (K-contiguous per half):
// lane<16: K kb..kb+15 with kb=0 ; lane>=16: kb=16.
static __device__ __forceinline__ v16bf load_fragA(const __bf16* rowptr, int kb) {
    v8bf lo = *(const v8bf*)(rowptr + kb);
    v8bf hi = *(const v8bf*)(rowptr + kb + 8);
    return __builtin_shufflevector(lo, hi, 0,1,2,3,4,5,6,7,8,9,10,11,12,13,14,15);
}

// B-operand fragment (K-contiguous per half): lane<16: K 0..15, lane>=16: K 16..31.
static __device__ __forceinline__ v16bf load_fragB(const __bf16* colptr, int hb) {
    v8bf lo = *(const v8bf*)(colptr + hb);
    v8bf hi = *(const v8bf*)(colptr + hb + 8);
    return __builtin_shufflevector(lo, hi, 0,1,2,3,4,5,6,7,8,9,10,11,12,13,14,15);
}

// ---------------------------------------------------------------------------
// Kernel 2: fused flash attention (transposed-S form) + gamma*out + residual.
// One wave owns a 16-column tile of S^T (16 attention rows i); block = 8 waves.
// S^T = k-tile(A) x q-tile(B)  -> softmax dim j lives on VGPR axis (in-lane).
// O^T = v-tile(A) x P^T(B)     -> P^T built in-registers (8 permlane swaps).
// ---------------------------------------------------------------------------
__global__ void __launch_bounds__(256)
cwca_attn_kernel(const __bf16* __restrict__ qT,
                 const __bf16* __restrict__ kT,
                 const __bf16* __restrict__ vM,
                 const float*  __restrict__ x1,
                 const float*  __restrict__ gamma,
                 float* __restrict__ out)
{
    const int lane = threadIdx.x & 31;
    const int wave = threadIdx.x >> 5;
    const int b    = blockIdx.y;
    const int i0   = blockIdx.x * 128 + wave * 16;

    const int laneN = lane & 15;        // this lane's attention row i (N-dim of S^T)
    const int half  = lane >> 4;
    const int kbase = half ? 16 : 0;    // A-layout K sub-offset
    const int hbase = half ? 16 : 0;    // B-layout K sub-offset

    // ---- B fragments: q columns i = i0+laneN, K = c (0..63, two fragments) ----
    const __bf16* qcol = qT + ((size_t)b * NPIX + i0 + laneN) * CCH;
    v16bf bq0 = load_fragB(qcol,      hbase);
    v16bf bq1 = load_fragB(qcol + 32, hbase);

    // ---- per-lane online softmax state (one i per lane) ----
    float m = -3.0e38f, l = 0.0f;
    v8f O0 = {}, O1 = {}, O2 = {}, O3 = {};

    const __bf16* kB = kT + (size_t)b * NPIX * CCH;
    const __bf16* vB = vM + (size_t)b * CCH * NPIX;
    // v A-operand row pointers: row c = chunk*16 + laneN, K = j contiguous
    const __bf16* vr0 = vB + (size_t)(laneN)      * NPIX;
    const __bf16* vr1 = vB + (size_t)(laneN + 16) * NPIX;
    const __bf16* vr2 = vB + (size_t)(laneN + 32) * NPIX;
    const __bf16* vr3 = vB + (size_t)(laneN + 48) * NPIX;

    for (int j0 = 0; j0 < NPIX; j0 += 32) {
        // ---- A fragments: k rows j0..j0+31 (two 16-row tiles, K=c=64) ----
        const __bf16* krow0 = kB + (size_t)(j0 + laneN) * CCH;
        const __bf16* krow1 = krow0 + (size_t)16 * CCH;
        v16bf ak00 = load_fragA(krow0,      kbase);
        v16bf ak01 = load_fragA(krow0 + 32, kbase);
        v16bf ak10 = load_fragA(krow1,      kbase);
        v16bf ak11 = load_fragA(krow1 + 32, kbase);

        // ---- S^T tiles: rows j (VGPR axis), cols i (lane axis) ----
        v8f s0 = {}, s1 = {};
        s0 = __builtin_amdgcn_wmma_f32_16x16x32_bf16(false, ak00, false, bq0, (short)0, s0, false, false);
        s0 = __builtin_amdgcn_wmma_f32_16x16x32_bf16(false, ak01, false, bq1, (short)0, s0, false, false);
        s1 = __builtin_amdgcn_wmma_f32_16x16x32_bf16(false, ak10, false, bq0, (short)0, s1, false, false);
        s1 = __builtin_amdgcn_wmma_f32_16x16x32_bf16(false, ak11, false, bq1, (short)0, s1, false, false);

        // ---- online softmax over j: in-lane reduce + one cross-half exchange ----
        float t = fmaxf(s0[0], s1[0]);
#pragma unroll
        for (int r = 1; r < 8; ++r) t = fmaxf(t, fmaxf(s0[r], s1[r]));
        t = fmaxf(t, xor16(t));
        float nm   = fmaxf(m, t);
        float corr = __expf(m - nm);
        m = nm;

        float e0[8], e1[8];
        float rs = 0.0f;
#pragma unroll
        for (int r = 0; r < 8; ++r) {
            e0[r] = __expf(s0[r] - nm);
            e1[r] = __expf(s1[r] - nm);
            rs += e0[r] + e1[r];
        }
        rs += xor16(rs);
        l = l * corr + rs;
        O0 *= corr; O1 *= corr; O2 *= corr; O3 *= corr;

        // ---- build P^T B-fragment in registers ----
        // lane<16 holds j {0..7} (own e0) and needs {8..15} from partner (its e0);
        // lane>=16 holds j {24..31} (own e1) and needs {16..23} from partner (its e1).
        v16bf bp;
#pragma unroll
        for (int r = 0; r < 8; ++r) {
            float send = half ? e0[r] : e1[r];
            float recv = xor16(send);
            float lo   = half ? recv  : e0[r];   // K = hbase + r
            float hi   = half ? e1[r] : recv;    // K = hbase + 8 + r
            bp[r]     = f2bf(lo);
            bp[8 + r] = f2bf(hi);
        }

        // ---- O^T += v x P^T : A = v tiles (M=c, K=j), 4 c-chunks ----
        v16bf av0 = load_fragA(vr0 + j0, kbase);
        v16bf av1 = load_fragA(vr1 + j0, kbase);
        v16bf av2 = load_fragA(vr2 + j0, kbase);
        v16bf av3 = load_fragA(vr3 + j0, kbase);

        O0 = __builtin_amdgcn_wmma_f32_16x16x32_bf16(false, av0, false, bp, (short)0, O0, false, false);
        O1 = __builtin_amdgcn_wmma_f32_16x16x32_bf16(false, av1, false, bp, (short)0, O1, false, false);
        O2 = __builtin_amdgcn_wmma_f32_16x16x32_bf16(false, av2, false, bp, (short)0, O2, false, false);
        O3 = __builtin_amdgcn_wmma_f32_16x16x32_bf16(false, av3, false, bp, (short)0, O3, false, false);
    }

    // ---- epilogue: out[b,c,i] = gamma * O^T[c,i]/l[i] + x1[b,c,i] ----
    const float g    = gamma[0];
    const float gl   = g / l;            // per-lane (per-i) scale
    const int   i    = i0 + laneN;
    const size_t base = (size_t)b * CCH * NPIX + i;

#pragma unroll
    for (int r = 0; r < 8; ++r) {
        int c0 = r + (half << 3);        // O VGPR r -> row c0 (+8 for upper half)
        size_t o0 = base + (size_t)(c0)      * NPIX;
        size_t o1 = base + (size_t)(c0 + 16) * NPIX;
        size_t o2 = base + (size_t)(c0 + 32) * NPIX;
        size_t o3 = base + (size_t)(c0 + 48) * NPIX;
        out[o0] = gl * O0[r] + x1[o0];
        out[o1] = gl * O1[r] + x1[o1];
        out[o2] = gl * O2[r] + x1[o2];
        out[o3] = gl * O3[r] + x1[o3];
    }
}

extern "C" void kernel_launch(void* const* d_in, const int* in_sizes, int n_in,
                              void* d_out, int out_size, void* d_ws, size_t ws_size,
                              hipStream_t stream) {
    const float* x1    = (const float*)d_in[0];
    const float* x2    = (const float*)d_in[1];
    const float* wq    = (const float*)d_in[2];
    const float* bq    = (const float*)d_in[3];
    const float* wk    = (const float*)d_in[4];
    const float* bk    = (const float*)d_in[5];
    const float* wv    = (const float*)d_in[6];
    const float* bv    = (const float*)d_in[7];
    const float* gamma = (const float*)d_in[8];
    float* out = (float*)d_out;

    const size_t elems = (size_t)BATCH * NPIX * CCH;       // 1M elements
    __bf16* qT = (__bf16*)d_ws;
    __bf16* kT = qT + elems;
    __bf16* vM = kT + elems;

    // projections: B*C*N threads
    {
        int total = BATCH * CCH * NPIX;
        dim3 grid(total / 256), block(256);
        hipLaunchKernelGGL(cwca_proj_kernel, grid, block, 0, stream,
                           x1, x2, wq, bq, wk, bk, wv, bv, qT, kT, vM);
    }
    // fused flash attention + residual
    {
        dim3 grid(NPIX / 128, BATCH), block(256);
        hipLaunchKernelGGL(cwca_attn_kernel, grid, block, 0, stream,
                           qT, kT, vM, x1, gamma, out);
    }
}
